// FeedzaiTrainAsync_54296976556061
// MI455X (gfx1250) — matrix operand order repaired and checked by reading, not compiled, not run on hardware
//
#include <hip/hip_runtime.h>
#include <math.h>

// ---------------------------------------------------------------------------
// Problem constants (from reference)
// ---------------------------------------------------------------------------
#define BATCH 4096
#define TSTEPS 200
#define FEAT 18
#define FPAD 32          // F padded to one bf16-WMMA K chunk
#define UNITS 128
#define NIDS 1000
#define G3U (3 * UNITS)  // 384 gate columns
#define ROWS 32          // batch rows per block (2 M-tiles; 128 blocks/step)
#define MT (ROWS / 16)   // M-tiles per block
#define NWAVES 8         // 256 threads / wave32

typedef __attribute__((ext_vector_type(16))) __bf16 v16bf;
typedef __attribute__((ext_vector_type(8)))  __bf16 v8bf;
typedef __attribute__((ext_vector_type(8)))  float  v8f;

// ---------------------------------------------------------------------------
// Fragment helpers (bf16 16x16x32 WMMA, wave32; this code's contiguous k order, same for A and B)
// A (16x32): lane<16 -> row=lane, K = {0..15}; lane>=16 -> row=lane-16,
//            K = {16..31}.  Two aligned 16B LDS loads per lane.
// B (32x16): stored pre-swizzled fragment-linear, 32B contiguous per lane.
// ---------------------------------------------------------------------------
__device__ __forceinline__ v16bf load_a_frag(const __bf16* rowptr, int half) {
  const v8bf lo = *(const v8bf*)(rowptr + 16 * half);
  const v8bf hi = *(const v8bf*)(rowptr + 16 * half + 8);
  v16bf a;
#pragma unroll
  for (int e = 0; e < 8; ++e) { a[e] = lo[e]; a[8 + e] = hi[e]; }
  return a;
}

__device__ __forceinline__ v16bf load_b_frag(const __bf16* p) {
  const v8bf lo = *(const v8bf*)(p);
  const v8bf hi = *(const v8bf*)(p + 8);
  v16bf b;
#pragma unroll
  for (int e = 0; e < 8; ++e) { b[e] = lo[e]; b[8 + e] = hi[e]; }
  return b;
}

__device__ __forceinline__ float hard_sigmoid(float x) {
  return fminf(fmaxf(0.2f * x + 0.5f, 0.f), 1.f);
}

// ---------------------------------------------------------------------------
// Weight conversion: f32 row-major [K][384] -> bf16 B-fragment-linear layout
//   frag[ntile][ktile][lane][j],  k = ktile*32 + j + 16*(lane>>4),
//   n = ntile*16 + (lane&15).  Wk is K=18 zero-padded to 32.
// ---------------------------------------------------------------------------
__global__ void convert_weights(const float* __restrict__ kern,
                                const float* __restrict__ rker,
                                __bf16* __restrict__ wkfrag,
                                __bf16* __restrict__ rfrag) {
  int idx = blockIdx.x * blockDim.x + threadIdx.x;
  if (idx < 24 * 32 * 16) {  // Wk fragments: 24 ntiles x 1 ktile
    int j = idx & 15, lane = (idx >> 4) & 31, n = idx >> 9;
    int k = j + 16 * (lane >> 4);
    int col = n * 16 + (lane & 15);
    float v = (k < FEAT) ? kern[k * G3U + col] : 0.f;
    wkfrag[idx] = (__bf16)v;
  }
  if (idx < 24 * 4 * 32 * 16) {  // R fragments: 24 ntiles x 4 ktiles
    int j = idx & 15, lane = (idx >> 4) & 31, kt = (idx >> 9) & 3, n = idx >> 11;
    int k = kt * 32 + j + 16 * (lane >> 4);
    int col = n * 16 + (lane & 15);
    rfrag[idx] = (__bf16)rker[k * G3U + col];
  }
}

// ---------------------------------------------------------------------------
// One GRU timestep for 32 batch rows per block (128 blocks/step).  Operands
// staged in ~84 KB dynamic LDS -> two blocks co-resident per 320 KB WGP for
// latency hiding.  Gate GEMMs on bf16 WMMA, f32 accumulation; B-fragments
// held in registers across the M-tiles.  Also records per-id winner row
// (atomicMax) for the deterministic last-row-wins scatter.
// ---------------------------------------------------------------------------
__global__ void gru_step(const float* __restrict__ inputs, int t,
                         const float* __restrict__ table,
                         const __bf16* __restrict__ wkfrag,
                         const __bf16* __restrict__ rfrag,
                         const float* __restrict__ bias,
                         float* __restrict__ hbuf,
                         int* __restrict__ winner) {
  extern __shared__ char smem[];
  float*  sh_h  = (float*)smem;                       // [32][128] f32   16 KB
  float*  sh_gz = sh_h  + ROWS * UNITS;               // gate z cols     16 KB
  float*  sh_gr = sh_gz + ROWS * UNITS;               // gate r cols     16 KB
  float*  sh_gh = sh_gr + ROWS * UNITS;               // gate hh cols    16 KB
  __bf16* sh_hb = (__bf16*)(sh_gh + ROWS * UNITS);    // h bf16           8 KB
  __bf16* sh_x  = sh_hb + ROWS * UNITS;               // x bf16 K=32      2 KB
  __bf16* sh_rh = sh_x  + ROWS * FPAD;                // r*h bf16         8 KB
  int*    sh_id = (int*)(sh_rh + ROWS * UNITS);       // ids             128 B

  const int tid  = threadIdx.x;
  const int row0 = blockIdx.x * ROWS;
  const int wave = tid >> 5, lane = tid & 31;
  const int half = lane >> 4, ln = lane & 15;

  // ---- Phase 1: load x_t rows (bf16, zero-pad K to 32), ids, winner ------
  for (int i = tid; i < ROWS * FPAD; i += 256) {
    int r = i >> 5, c = i & 31;
    const float* src = inputs + ((size_t)(row0 + r) * TSTEPS + t) * FEAT;
    float v = 0.f;
    if (c < FEAT) v = src[c];
    sh_x[r * FPAD + c] = (__bf16)v;
    if (c == 0) {
      int id = (int)v;
      sh_id[r] = id;
      atomicMax(&winner[id], row0 + r);  // deterministic last-row-wins
      if (t + 1 < TSTEPS) {              // warm L2 for next timestep's x row
        __builtin_prefetch(src + FEAT, 0, 0);
        __builtin_prefetch(src + FEAT + 16, 0, 0);
      }
    }
  }
  __syncthreads();

  // ---- gather h from shared-state table (f32 + bf16 copies) ---------------
  for (int i = tid; i < ROWS * UNITS; i += 256) {
    int r = i >> 7, c = i & 127;
    float hv = table[sh_id[r] * UNITS + c];
    sh_h[r * UNITS + c]  = hv;
    sh_hb[r * UNITS + c] = (__bf16)hv;
  }
  __syncthreads();

  // ---- Phase 2: gates = bias + x@Wk (+ h@R for first 256 cols) ------------
  // Each wave owns n-tiles {wave, wave+8, wave+16}; B-fragments loaded once
  // per n-tile and reused across the M-tiles (held in registers).
#pragma unroll 1
  for (int ni = 0; ni < 3; ++ni) {
    const int n = wave + NWAVES * ni;
    const v16bf bx = load_b_frag(wkfrag + (size_t)(n * 32 + lane) * 16);
    v16bf bh[4];
    const bool has_h = (n < 16);
    if (has_h) {
#pragma unroll
      for (int kk = 0; kk < 4; ++kk)
        bh[kk] = load_b_frag(rfrag + (size_t)((n * 4 + kk) * 32 + lane) * 16);
    }
    const float bcol = bias[n * 16 + ln];
    float* gdst = (n < 8) ? sh_gz : (n < 16) ? sh_gr : sh_gh;
    const int gc = (n & 7) * 16 + ln;
#pragma unroll
    for (int m = 0; m < MT; ++m) {
      v16bf ax = load_a_frag(sh_x + (m * 16 + ln) * FPAD, half);
      v8f acc = {};
      acc = __builtin_amdgcn_wmma_f32_16x16x32_bf16(
          false, ax, false, bx, (short)0, acc, false, false);
      if (has_h) {
#pragma unroll
        for (int kk = 0; kk < 4; ++kk) {
          v16bf ah = load_a_frag(sh_hb + (m * 16 + ln) * UNITS + kk * 32, half);
          acc = __builtin_amdgcn_wmma_f32_16x16x32_bf16(
              false, ah, false, bh[kk], (short)0, acc, false, false);
        }
      }
      const int rbase = m * 16 + half * 8;
#pragma unroll
      for (int v = 0; v < 8; ++v)
        gdst[(rbase + v) * UNITS + gc] = acc[v] + bcol;
    }
  }
  __syncthreads();

  // ---- Phase 3: z, r, r*h (bf16) ------------------------------------------
  for (int i = tid; i < ROWS * UNITS; i += 256) {
    int r = i >> 7, c = i & 127;
    float z  = hard_sigmoid(sh_gz[r * UNITS + c]);
    float rr = hard_sigmoid(sh_gr[r * UNITS + c]);
    sh_gz[r * UNITS + c] = z;  // keep z for h_new
    sh_rh[r * UNITS + c] = (__bf16)(rr * sh_h[r * UNITS + c]);
  }
  __syncthreads();

  // ---- Phase 4: hh = tanh(gh + (r*h)@R[:,256:]);  h_new = z*h + (1-z)*hh --
  // One n-tile per wave (N=128 -> 8 tiles); B held in regs across M-tiles.
  {
    const int n = wave;
    v16bf bh[4];
#pragma unroll
    for (int kk = 0; kk < 4; ++kk)
      bh[kk] = load_b_frag(rfrag + (size_t)(((16 + n) * 4 + kk) * 32 + lane) * 16);
    const int col = n * 16 + ln;
#pragma unroll
    for (int m = 0; m < MT; ++m) {
      const int rbase = m * 16 + half * 8;
      v8f acc;
#pragma unroll
      for (int v = 0; v < 8; ++v) acc[v] = sh_gh[(rbase + v) * UNITS + col];
#pragma unroll
      for (int kk = 0; kk < 4; ++kk) {
        v16bf a = load_a_frag(sh_rh + (m * 16 + ln) * UNITS + kk * 32, half);
        acc = __builtin_amdgcn_wmma_f32_16x16x32_bf16(
            false, a, false, bh[kk], (short)0, acc, false, false);
      }
#pragma unroll
      for (int v = 0; v < 8; ++v) {
        int r = rbase + v;
        float hh = tanhf(acc[v]);
        float z = sh_gz[r * UNITS + col];
        float h = sh_h[r * UNITS + col];
        sh_h[r * UNITS + col] = z * h + (1.f - z) * hh;  // h_new in place
      }
    }
  }
  __syncthreads();

  // ---- Phase 5: emit h_new -------------------------------------------------
  for (int i = tid; i < ROWS * UNITS; i += 256) {
    int r = i >> 7, c = i & 127;
    hbuf[(size_t)(row0 + r) * UNITS + c] = sh_h[r * UNITS + c];
  }
}

// ---------------------------------------------------------------------------
// Winner-buffer init (both ping-pong buffers, once per launch).
// ---------------------------------------------------------------------------
__global__ void winner_init(int* __restrict__ winner) {
  int i = blockIdx.x * blockDim.x + threadIdx.x;
  if (i < 2 * 1024) winner[i] = -1;
}

// ---------------------------------------------------------------------------
// Deterministic scatter: winners (max batch row per id) write the table.
// Also resets the other ping-pong winner buffer for the next timestep.
// ---------------------------------------------------------------------------
__global__ void scatter_update(const float* __restrict__ inputs, int t,
                               const int* __restrict__ winner_cur,
                               int* __restrict__ winner_next,
                               const float* __restrict__ hbuf,
                               float* __restrict__ table) {
  int idx = blockIdx.x * blockDim.x + threadIdx.x;
  if (idx < NIDS) winner_next[idx] = -1;
  if (idx < BATCH * UNITS) {
    int row = idx >> 7, c = idx & 127;
    int id = (int)inputs[((size_t)row * TSTEPS + t) * FEAT];
    if (winner_cur[id] == row)
      table[id * UNITS + c] = hbuf[(size_t)row * UNITS + c];
  }
}

// ---------------------------------------------------------------------------
// Final MLP: out = sigmoid(relu(h_last@w1 + b1) @ w2 + b2).  Tiny; fp32 VALU.
// ---------------------------------------------------------------------------
__global__ void mlp_head(const float* __restrict__ hbuf,
                         const float* __restrict__ w1,
                         const float* __restrict__ b1,
                         const float* __restrict__ w2,
                         const float* __restrict__ b2,
                         float* __restrict__ out) {
  __shared__ float sh[UNITS];
  __shared__ float sd[64];
  int row = blockIdx.x, tid = threadIdx.x;  // 64 threads
  sh[tid]      = hbuf[(size_t)row * UNITS + tid];
  sh[tid + 64] = hbuf[(size_t)row * UNITS + 64 + tid];
  __syncthreads();
  float a = b1[tid];
#pragma unroll 4
  for (int k = 0; k < UNITS; ++k) a += sh[k] * w1[k * 64 + tid];
  sd[tid] = fmaxf(a, 0.f) * w2[tid];
  __syncthreads();
  if (tid == 0) {
    float s = b2[0];
#pragma unroll
    for (int j = 0; j < 64; ++j) s += sd[j];
    out[row] = 1.f / (1.f + expf(-s));
  }
}

// ---------------------------------------------------------------------------
// Launch
// ---------------------------------------------------------------------------
extern "C" void kernel_launch(void* const* d_in, const int* in_sizes, int n_in,
                              void* d_out, int out_size, void* d_ws, size_t ws_size,
                              hipStream_t stream) {
  (void)in_sizes; (void)n_in; (void)out_size; (void)ws_size;
  const float* inputs  = (const float*)d_in[0];
  const float* kern    = (const float*)d_in[1];
  const float* rker    = (const float*)d_in[2];
  const float* bias    = (const float*)d_in[3];
  const float* shared0 = (const float*)d_in[4];
  const float* w1 = (const float*)d_in[5];
  const float* b1 = (const float*)d_in[6];
  const float* w2 = (const float*)d_in[7];
  const float* b2 = (const float*)d_in[8];
  float* out = (float*)d_out;

  char* ws = (char*)d_ws;
  float*  table  = (float*)ws;                      // 512000 B (mutable copy)
  __bf16* wkfrag = (__bf16*)(ws + 512000);          //  24576 B
  __bf16* rfrag  = (__bf16*)(ws + 536576);          //  98304 B
  float*  hbuf   = (float*)(ws + 634880);           // 2097152 B
  int*    winner = (int*)(ws + 2732032);            // 2 x 1024 ints (ping-pong)

  // Fresh mutable copy of the shared-state table every call (determinism;
  // never mutate d_in).
  hipMemcpyAsync(table, shared0, (size_t)NIDS * UNITS * sizeof(float),
                 hipMemcpyDeviceToDevice, stream);
  convert_weights<<<192, 256, 0, stream>>>(kern, rker, wkfrag, rfrag);
  winner_init<<<8, 256, 0, stream>>>(winner);

  constexpr size_t SMEM =
      (size_t)4 * ROWS * UNITS * sizeof(float) +      // h, gz, gr, gh
      (size_t)ROWS * UNITS * sizeof(__bf16) +         // hb
      (size_t)ROWS * FPAD  * sizeof(__bf16) +         // x
      (size_t)ROWS * UNITS * sizeof(__bf16) +         // rh
      (size_t)ROWS * sizeof(int);                     // ids  => 84096 B

  for (int t = 0; t < TSTEPS; ++t) {
    int* wcur = winner + (t & 1) * 1024;
    int* wnxt = winner + ((t + 1) & 1) * 1024;
    gru_step<<<BATCH / ROWS, 256, SMEM, stream>>>(inputs, t, table, wkfrag,
                                                  rfrag, bias, hbuf, wcur);
    scatter_update<<<(BATCH * UNITS) / 256, 256, 0, stream>>>(
        inputs, t, wcur, wnxt, hbuf, table);
  }
  mlp_head<<<BATCH, 64, 0, stream>>>(hbuf, w1, b1, w2, b2, out);
}
